// FFTConvergedInhibition_80848464380249
// MI455X (gfx1250) — matrix operand order repaired and checked by reading, not compiled, not run on hardware
//
#include <hip/hip_runtime.h>
#include <hip/hip_bf16.h>
#include <math.h>

// ---------------------------------------------------------------------------
// FFT "converged inhibition" == channel-circulant solve:
//   y[n,:,h,w] = (I - K)^-1 x[n,:,h,w]  ==  M @ x  with M 512x512 circulant.
// Pipeline: (1) impulse response m via direct DFT, (2) materialize M as
// bf16 hi/lo, (3) tiled WMMA GEMM M(512x512) x X(512x50176), fp32-class
// accuracy via 3-product bf16 split accumulated in f32.
// ---------------------------------------------------------------------------

typedef __attribute__((ext_vector_type(16))) __bf16          v16bf;
typedef __attribute__((ext_vector_type(8)))  float           v8f;
typedef __attribute__((ext_vector_type(8)))  unsigned short  u16x8;
typedef __attribute__((ext_vector_type(16))) unsigned short  u16x16;

#define C_CH   512
#define KSCOPE 27
#define N_IMG  16
#define HW     3136           // 56*56
#define PB     16             // pixels per workgroup (one 16-wide N tile)
#define PITCH  520            // LDS row pitch (bf16 elems), 520*2 % 16 == 0

__device__ __forceinline__ unsigned short f32_to_bf16(float f) {
    unsigned int u = __float_as_uint(f);
    unsigned int r = 0x7FFFu + ((u >> 16) & 1u);   // round-to-nearest-even
    return (unsigned short)((u + r) >> 16);
}
__device__ __forceinline__ float bf16_to_f32(unsigned short h) {
    return __uint_as_float(((unsigned int)h) << 16);
}

// --- Kernel 1: m[c] = (1/C) * sum_f Re( e^{+i2pi f c/C} / FK[f] ) ----------
// FK[f] = DFT of (delta - roll(pad(filter), -13)). Support = 27 taps at
// offsets t = s-13 (mod 512); center tap of the filter is already zero.
__global__ void build_impulse(const float* __restrict__ filt,
                              float* __restrict__ m) {
    __shared__ float fr_s[C_CH];
    __shared__ float fi_s[C_CH];
    const int f = threadIdx.x;
    const double w0 = 6.283185307179586476925286766559 / (double)C_CH;

    double fr = 1.0, fi = 0.0;                 // delta at t=0 contributes 1
    for (int s = 0; s < KSCOPE; ++s) {
        int t = (s - (KSCOPE / 2)) & (C_CH - 1);
        double a  = (double)((f * t) & (C_CH - 1)) * w0;  // exact angle index
        double fv = (double)filt[s];
        fr -= fv * cos(a);                      // -(filt) * e^{-ia}, real
        fi += fv * sin(a);                      //                  imag
    }
    fr_s[f] = (float)fr;
    fi_s[f] = (float)fi;
    __syncthreads();

    const int c = f;
    double acc = 0.0;
    for (int q = 0; q < C_CH; ++q) {
        double qr = (double)fr_s[q], qi = (double)fi_s[q];
        double a  = (double)((q * c) & (C_CH - 1)) * w0;
        // Re( e^{ia} / (qr + i qi) ) = (cos*qr + sin*qi) / |FK|^2
        acc += (cos(a) * qr + sin(a) * qi) / (qr * qr + qi * qi);
    }
    m[c] = (float)(acc / (double)C_CH);
}

// --- Kernel 2: M[c][j] = m[(c-j) mod 512], split into bf16 hi + lo --------
__global__ void build_M(const float* __restrict__ m,
                        unsigned short* __restrict__ Mhi,
                        unsigned short* __restrict__ Mlo) {
    int idx = blockIdx.x * blockDim.x + threadIdx.x;   // 512*512 total
    int c = idx >> 9, j = idx & (C_CH - 1);
    float v = m[(c - j) & (C_CH - 1)];
    unsigned short h = f32_to_bf16(v);
    unsigned short l = f32_to_bf16(v - bf16_to_f32(h));
    Mhi[idx] = h;
    Mlo[idx] = l;
}

// --- Kernel 3: Y[n] = M @ X[n] via v_wmma_f32_16x16x32_bf16 ----------------
// Block: 256 threads = 8 wave32s. Block covers all 512 output channels for a
// 16-pixel column block, so X is read from HBM exactly once. M (1 MB bf16)
// is L2-resident (192 MB L2). Wave w -> channels [w*64, w*64+64) = 4 tiles.
__global__ __launch_bounds__(256) void inhib_gemm(
    const float* __restrict__ x,
    const unsigned short* __restrict__ Mhi,
    const unsigned short* __restrict__ Mlo,
    float* __restrict__ y) {
    __shared__ __align__(16) unsigned short sXh[PB * PITCH];
    __shared__ __align__(16) unsigned short sXl[PB * PITCH];

    const int n   = blockIdx.y;
    const int p0  = blockIdx.x * PB;
    const int tid = threadIdx.x;

    // Stage X[n, :, p0:p0+16] -> LDS, transposed [p][j], bf16 hi/lo split.
    const float* xn = x + (size_t)n * C_CH * HW + p0;
    for (int e = tid; e < C_CH * PB; e += 256) {
        int j = e >> 4, p = e & 15;               // p fastest -> coalesced
        float v = xn[(size_t)j * HW + p];
        unsigned short h = f32_to_bf16(v);
        unsigned short l = f32_to_bf16(v - bf16_to_f32(h));
        sXh[p * PITCH + j] = h;
        sXl[p * PITCH + j] = l;
    }
    __syncthreads();

    const int wave = tid >> 5;        // wave32 (gfx1250)
    const int lane = tid & 31;
    const int half = lane >> 4;
    const int l16  = lane & 15;
    const int c0   = wave * 64;

    v8f acc[4] = {v8f{}, v8f{}, v8f{}, v8f{}};

    const unsigned short* bBaseH = &sXh[l16 * PITCH];
    const unsigned short* bBaseL = &sXl[l16 * PITCH];

    for (int ks = 0; ks < 16; ++ks) {
        const int j0 = ks * 32;

        // B fragment (32x16): lanes 0-15 hold K={j0..j0+7, j0+16..j0+23} of column N=l16,
        // lanes 16-31 shifted by +8 (same K order as A).
        const int bo = j0 + 8 * half;
        u16x8 b0h = *(const u16x8*)(bBaseH + bo);
        u16x8 b1h = *(const u16x8*)(bBaseH + bo + 16);
        u16x8 b0l = *(const u16x8*)(bBaseL + bo);
        u16x8 b1l = *(const u16x8*)(bBaseL + bo + 16);
        u16x16 bhu, blu;
#pragma unroll
        for (int i = 0; i < 8; ++i) {
            bhu[i] = b0h[i]; bhu[i + 8] = b1h[i];
            blu[i] = b0l[i]; blu[i + 8] = b1l[i];
        }
        v16bf Bh = __builtin_bit_cast(v16bf, bhu);
        v16bf Bl = __builtin_bit_cast(v16bf, blu);

#pragma unroll
        for (int t = 0; t < 4; ++t) {
            // A fragment (16x32): lanes 0-15 K={j0..j0+7, j0+16..j0+23},
            // lanes 16-31 shifted by +8 (ISA 05_wmma 16-bit A layout).
            const int c = c0 + t * 16 + l16;
            const unsigned short* arH = Mhi + (size_t)c * C_CH + j0 + 8 * half;
            const unsigned short* arL = Mlo + (size_t)c * C_CH + j0 + 8 * half;
            u16x8 a0h = *(const u16x8*)(arH);
            u16x8 a1h = *(const u16x8*)(arH + 16);
            u16x8 a0l = *(const u16x8*)(arL);
            u16x8 a1l = *(const u16x8*)(arL + 16);
            u16x16 ahu, alu;
#pragma unroll
            for (int i = 0; i < 8; ++i) {
                ahu[i] = a0h[i]; ahu[i + 8] = a1h[i];
                alu[i] = a0l[i]; alu[i + 8] = a1l[i];
            }
            v16bf Ah = __builtin_bit_cast(v16bf, ahu);
            v16bf Al = __builtin_bit_cast(v16bf, alu);

            // fp32-class accuracy: Ah*Bh + Ah*Bl + Al*Bh (drop lo*lo ~2^-32)
            acc[t] = __builtin_amdgcn_wmma_f32_16x16x32_bf16(
                false, Ah, false, Bh, (short)0, acc[t], false, false);
            acc[t] = __builtin_amdgcn_wmma_f32_16x16x32_bf16(
                false, Ah, false, Bl, (short)0, acc[t], false, false);
            acc[t] = __builtin_amdgcn_wmma_f32_16x16x32_bf16(
                false, Al, false, Bh, (short)0, acc[t], false, false);
        }
    }

    // C/D layout (ISA 7.12.2): VGPR r, lanes 0-15 -> M=r, lanes 16-31 -> M=8+r,
    // N = lane%16. Store coalesced (consecutive l16 -> consecutive pixels).
    float* yn = y + (size_t)n * C_CH * HW + p0 + l16;
#pragma unroll
    for (int t = 0; t < 4; ++t) {
#pragma unroll
        for (int r = 0; r < 8; ++r) {
            int c = c0 + t * 16 + 8 * half + r;
            yn[(size_t)c * HW] = acc[t][r];
        }
    }
}

extern "C" void kernel_launch(void* const* d_in, const int* in_sizes, int n_in,
                              void* d_out, int out_size, void* d_ws, size_t ws_size,
                              hipStream_t stream) {
    const float* x    = (const float*)d_in[0];   // activations [16,512,56,56]
    const float* filt = (const float*)d_in[1];   // inhibition_filter [27]
    float* out = (float*)d_out;

    // Workspace layout (needs ~1.06 MB):
    //   [0,2KB)        m[512] f32 impulse response
    //   [4KB, +512KB)  Mhi bf16 512x512
    //   [+512KB, ...)  Mlo bf16 512x512
    char* ws = (char*)d_ws;
    float* m = (float*)ws;
    unsigned short* Mhi = (unsigned short*)(ws + 4096);
    unsigned short* Mlo = (unsigned short*)(ws + 4096 + (size_t)C_CH * C_CH * 2);

    build_impulse<<<1, C_CH, 0, stream>>>(filt, m);
    build_M<<<(C_CH * C_CH) / 256, 256, 0, stream>>>(m, Mhi, Mlo);

    dim3 grid(HW / PB, N_IMG);   // 196 x 16 blocks, 256 threads (8 wave32s)
    inhib_gemm<<<grid, 256, 0, stream>>>(x, Mhi, Mlo, out);
}
